// LinearAttentionBlock_4389456577366
// MI455X (gfx1250) — matrix operand order repaired and checked by reading, not compiled, not run on hardware
//
#include <hip/hip_runtime.h>

// ---------------------------------------------------------------------------
// CDNA5 / gfx1250 causal linear attention block, bf16 WMMA (f32 accumulate).
// Single templated GEMM, double-buffered LDS, 128x64 block tile, 32x32 wave
// tile (8 waves). ALL B-operands are column-major [N,K] so every tile load is
// a contiguous b128 vector copy (weights pre-transposed once; V produced
// transposed by the projection GEMM's TRANSOUT epilogue).
// The k-loop barrier is a hand-rolled split barrier that only waits on DScnt,
// so prefetched global loads for tile t+1 stay in flight across the barrier
// and the WMMAs of tile t (the __syncthreads() fence would wait LOADcnt too).
// Pipeline:
//   1) x: f32->bf16 copy; weights: f32->bf16 tiled transpose
//   2) gemm: Q = phi(x WqT^T + bq), K = phi(x WkT^T + bk)        (bf16)
//      gemm: Vt = (x WvT^T + bv)^T                               (bf16, TRANSOUT)
//   3) gemm (MASKTRIL): S = tril(Q K^T)                          (bf16)
//   4) rowsum: den[b,r] = eps + sum_m S[b,r,m]                   (deterministic)
//   5) gemm (CAUSALK k-limit + ROWSCALE): A = (S V) / den        (bf16)
//   6) gemm: out = A Wo + bo                                     (f32)
// ---------------------------------------------------------------------------

typedef __bf16 bf16;
typedef __attribute__((ext_vector_type(16))) __bf16 v16bf;
typedef __attribute__((ext_vector_type(8)))  float  v8f;
typedef __attribute__((ext_vector_type(4)))  float  f32x4;

#define DMODEL   1024
#define DINNER   2048
#define NBATCH   4
#define SEQLEN   2048
#define EPSV     1e-6f

#define BM 128           // block rows
#define BN 64            // block cols
#define BK 32            // k-chunk
#define NTHREADS 256     // 8 waves: 4 row-waves x 2 col-waves, 32x32 per wave

struct BF16x16 { f32x4 lo, hi; };

__device__ __forceinline__ v16bf frag_ld(const bf16* p0, const bf16* p1) {
  BF16x16 t;
  t.lo = *(const f32x4*)p0;
  t.hi = *(const f32x4*)p1;
  return __builtin_bit_cast(v16bf, t);
}

__device__ __forceinline__ v8f wmma_bf16(v16bf a, v16bf b, v8f c) {
  return __builtin_amdgcn_wmma_f32_16x16x32_bf16(false, a, false, b,
                                                 (short)0, c, false, false);
}

// Workgroup barrier that orders only LDS traffic (DScnt), leaving prefetched
// global loads in flight. CDNA5 split barrier + split wait counters.
__device__ __forceinline__ void lds_barrier() {
  asm volatile("s_wait_dscnt 0x0\n\t"
               "s_barrier_signal -1\n\t"
               "s_barrier_wait -1" ::: "memory");
}

// ---------------------------------------------------------------------------
// out = epilogue(A[M,K] * BT[N,K]^T + bias)
//   ACT==1   : elu(x)+1 epilogue
//   OUTF32   : store f32, else bf16
//   CAUSALK  : k-loop limited to row0+BM (A is lower-triangular scores)
//   ROWSCALE : multiply by 1/denom[row] before store
//   MASKTRIL : scores mode: skip blocks above diagonal (bx > 2*by+1 for
//              BM=2*BN) and zero out entries with col > row
//   TRANSOUT : store out[col][row] (leading dim ldo = rows)
// blockIdx.z = batch with element strides sA/sB/sOut/sDen.
// ---------------------------------------------------------------------------
template<int ACT, bool OUTF32, bool CAUSALK, bool ROWSCALE, bool MASKTRIL,
         bool TRANSOUT>
__global__ __launch_bounds__(NTHREADS)
void wmma_gemm(const bf16* __restrict__ A, const bf16* __restrict__ BT,
               const float* __restrict__ bias, const float* __restrict__ denom,
               void* __restrict__ out, int N, int K, int ldo,
               long long sA, long long sB, long long sOut, long long sDen)
{
  if (MASKTRIL && (int)blockIdx.x > 2 * (int)blockIdx.y + 1) return;

  __shared__ __align__(16) bf16 As[2][BM][BK];
  __shared__ __align__(16) bf16 BsT[2][BN][BK];

  const int bz = blockIdx.z;
  A  += (long long)bz * sA;
  BT += (long long)bz * sB;
  if (ROWSCALE) denom += (long long)bz * sDen;

  const int row0 = blockIdx.y * BM;
  const int col0 = blockIdx.x * BN;
  const int tid  = threadIdx.x;
  const int lane = tid & 31;
  const int wave = tid >> 5;
  const int wr   = wave >> 1;     // 0..3 : 32-row slab
  const int wc   = wave & 1;      // 0..1 : 32-col slab
  const int half = lane >> 4;
  const int l16  = lane & 15;

  v8f acc[2][2];
  #pragma unroll
  for (int i = 0; i < 2; ++i)
    #pragma unroll
    for (int j = 0; j < 2; ++j)
      acc[i][j] = (v8f){0.f,0.f,0.f,0.f,0.f,0.f,0.f,0.f};

  int kEnd = K;
  if (CAUSALK) { const int lim = row0 + BM; kEnd = (lim < K) ? lim : K; }
  const int nT = kEnd / BK;

  // cooperative-load geometry: every thread moves three 16B chunks per tile
  const int ar = tid >> 2;           // 0..63 : A rows ar, ar+64 ; BT row ar
  const int ac = (tid & 3) * 8;      // 8-element k chunk

  f32x4 aReg0, aReg1, bReg;

  auto gload = [&](int kb) {
    aReg0 = *(const f32x4*)&A[(long long)(row0 + ar) * K + kb + ac];
    aReg1 = *(const f32x4*)&A[(long long)(row0 + 64 + ar) * K + kb + ac];
    bReg  = *(const f32x4*)&BT[(long long)(col0 + ar) * K + kb + ac];
  };
  auto lstore = [&](int buf) {
    *(f32x4*)&As[buf][ar][ac]      = aReg0;
    *(f32x4*)&As[buf][64 + ar][ac] = aReg1;
    *(f32x4*)&BsT[buf][ar][ac]     = bReg;
  };
  auto compute = [&](int buf) {
    const bf16* a0 = &As[buf][wr * 32 + l16][0];
    const bf16* a1 = &As[buf][wr * 32 + 16 + l16][0];
    v16bf af0 = frag_ld(a0 + half * 8, a0 + 16 + half * 8);
    v16bf af1 = frag_ld(a1 + half * 8, a1 + 16 + half * 8);
    const bf16* b0 = &BsT[buf][wc * 32 + l16][0];
    const bf16* b1 = &BsT[buf][wc * 32 + 16 + l16][0];
    v16bf bq0 = frag_ld(b0 + half * 8, b0 + 16 + half * 8);
    v16bf bq1 = frag_ld(b1 + half * 8, b1 + 16 + half * 8);
    acc[0][0] = wmma_bf16(af0, bq0, acc[0][0]);
    acc[0][1] = wmma_bf16(af0, bq1, acc[0][1]);
    acc[1][0] = wmma_bf16(af1, bq0, acc[1][0]);
    acc[1][1] = wmma_bf16(af1, bq1, acc[1][1]);
  };

  // software-pipelined ping-pong: gload(t+1) issued before the barrier of t;
  // lds_barrier() waits only DScnt so those loads stay in flight until the
  // ds_store in lstore() consumes the staged registers.
  gload(0);
  lstore(0);
  for (int t = 0; t < nT; ++t) {
    const int cur = t & 1;
    if (t + 1 < nT) gload((t + 1) * BK);
    lds_barrier();
    compute(cur);
    if (t + 1 < nT) lstore(cur ^ 1);
  }

  #pragma unroll
  for (int i = 0; i < 2; ++i) {
    #pragma unroll
    for (int j = 0; j < 2; ++j) {
      const int gcol = col0 + wc * 32 + j * 16 + l16;
      const float bval = bias ? bias[gcol] : 0.f;
      #pragma unroll
      for (int r = 0; r < 8; ++r) {
        const int grow = row0 + wr * 32 + i * 16 + half * 8 + r;
        float v = acc[i][j][r] + bval;
        if (ACT == 1) v = (v > 0.f) ? (v + 1.f) : __expf(v);
        if (ROWSCALE) v = v / denom[grow];
        if (MASKTRIL && gcol > grow) v = 0.f;
        const long long oidx = TRANSOUT
            ? (long long)bz * sOut + (long long)gcol * ldo + grow
            : (long long)bz * sOut + (long long)grow * ldo + gcol;
        if (OUTF32) ((float*)out)[oidx] = v;
        else        ((bf16*)out)[oidx]  = (bf16)v;
      }
    }
  }
}

// ---------------------------------------------------------------------------
// Deterministic denominator: den[b,r] = eps + sum_m S[b,r,m].
// Written prefix for row r is (r/BM + 1)*BM columns (rest masked zero).
// ---------------------------------------------------------------------------
__global__ __launch_bounds__(NTHREADS)
void rowsum_kernel(const bf16* __restrict__ S, float* __restrict__ denom)
{
  const int r = blockIdx.x, b = blockIdx.y;
  const bf16* row = S + ((long long)b * SEQLEN + r) * SEQLEN;
  const int lim = (r / BM + 1) * BM;
  float s = 0.f;
  for (int c = threadIdx.x; c < lim; c += NTHREADS) s += (float)row[c];
  #pragma unroll
  for (int m = 16; m >= 1; m >>= 1) s += __shfl_xor(s, m, 32);
  __shared__ float red[NTHREADS / 32];
  if ((threadIdx.x & 31) == 0) red[threadIdx.x >> 5] = s;
  __syncthreads();
  if (threadIdx.x == 0) {
    float t = 0.f;
    #pragma unroll
    for (int i = 0; i < NTHREADS / 32; ++i) t += red[i];
    denom[(long long)b * SEQLEN + r] = t + EPSV;
  }
}

__global__ void f32_to_bf16_kernel(const float* __restrict__ in,
                                   bf16* __restrict__ out, long long n)
{
  long long i = (long long)blockIdx.x * blockDim.x + threadIdx.x;
  const long long stride = (long long)gridDim.x * blockDim.x;
  for (; i < n; i += stride) out[i] = (bf16)in[i];
}

// out[C][R] = (bf16) in[R][C] ; R, C multiples of 32. grid = (C/32, R/32).
__global__ __launch_bounds__(NTHREADS)
void transpose_f32_to_bf16_kernel(const float* __restrict__ in,
                                  bf16* __restrict__ out, int R, int C)
{
  __shared__ float tile[32][33];
  const int r0 = blockIdx.y * 32, c0 = blockIdx.x * 32;
  const int tx = threadIdx.x & 31, ty = threadIdx.x >> 5;  // 32 x 8 threads
  #pragma unroll
  for (int i = 0; i < 4; ++i)
    tile[ty + i * 8][tx] = in[(long long)(r0 + ty + i * 8) * C + c0 + tx];
  __syncthreads();
  #pragma unroll
  for (int i = 0; i < 4; ++i)
    out[(long long)(c0 + ty + i * 8) * R + r0 + tx] = (bf16)tile[tx][ty + i * 8];
}

// ---------------------------------------------------------------------------
extern "C" void kernel_launch(void* const* d_in, const int* in_sizes, int n_in,
                              void* d_out, int out_size, void* d_ws, size_t ws_size,
                              hipStream_t stream)
{
  (void)in_sizes; (void)n_in; (void)out_size; (void)ws_size;

  const float* x  = (const float*)d_in[0];
  const float* Wq = (const float*)d_in[1];
  const float* bq = (const float*)d_in[2];
  const float* Wk = (const float*)d_in[3];
  const float* bk = (const float*)d_in[4];
  const float* Wv = (const float*)d_in[5];
  const float* bv = (const float*)d_in[6];
  const float* Wo = (const float*)d_in[7];
  const float* bo = (const float*)d_in[8];

  const long long MTOK = (long long)NBATCH * SEQLEN;   // 8192 tokens

  unsigned char* wsb = (unsigned char*)d_ws;
  size_t off = 0;
  auto take = [&](size_t bytes) -> void* {
    void* p = wsb + off;
    off += (bytes + 255) & ~((size_t)255);
    return p;
  };

  bf16* xb   = (bf16*)take((size_t)MTOK * DMODEL * sizeof(bf16));
  bf16* wqT  = (bf16*)take((size_t)DMODEL * DINNER * sizeof(bf16)); // [DINNER][DMODEL]
  bf16* wkT  = (bf16*)take((size_t)DMODEL * DINNER * sizeof(bf16));
  bf16* wvT  = (bf16*)take((size_t)DMODEL * DINNER * sizeof(bf16));
  bf16* woT  = (bf16*)take((size_t)DINNER * DMODEL * sizeof(bf16)); // [DMODEL][DINNER]
  bf16* Qb   = (bf16*)take((size_t)MTOK * DINNER * sizeof(bf16));   // [tok][e]
  bf16* Kb   = (bf16*)take((size_t)MTOK * DINNER * sizeof(bf16));   // [tok][e]
  bf16* Vt   = (bf16*)take((size_t)MTOK * DINNER * sizeof(bf16));   // [b][e][m]
  bf16* Sb   = (bf16*)take((size_t)NBATCH * SEQLEN * SEQLEN * sizeof(bf16));
  bf16* Ab   = (bf16*)take((size_t)MTOK * DINNER * sizeof(bf16));   // [tok][e]
  float* den = (float*)take((size_t)MTOK * sizeof(float));

  const dim3 blk(NTHREADS);

  // 1) conversions: x plain copy, weights transposed to [N][K]
  f32_to_bf16_kernel<<<2048, 256, 0, stream>>>(x, xb, MTOK * DMODEL);
  const dim3 gtw(DINNER / 32, DMODEL / 32);   // W* are [DMODEL][DINNER]
  transpose_f32_to_bf16_kernel<<<gtw, blk, 0, stream>>>(Wq, wqT, DMODEL, DINNER);
  transpose_f32_to_bf16_kernel<<<gtw, blk, 0, stream>>>(Wk, wkT, DMODEL, DINNER);
  transpose_f32_to_bf16_kernel<<<gtw, blk, 0, stream>>>(Wv, wvT, DMODEL, DINNER);
  const dim3 gto(DMODEL / 32, DINNER / 32);   // Wo is [DINNER][DMODEL]
  transpose_f32_to_bf16_kernel<<<gto, blk, 0, stream>>>(Wo, woT, DINNER, DMODEL);

  // 2) projections: Q, K (elu+1, row-major out), V (TRANSOUT per batch)
  const dim3 gqk(DINNER / BN, (unsigned)(MTOK / BM), 1);
  wmma_gemm<1, false, false, false, false, false><<<gqk, blk, 0, stream>>>(
      xb, wqT, bq, nullptr, Qb, DINNER, DMODEL, DINNER, 0, 0, 0, 0);
  wmma_gemm<1, false, false, false, false, false><<<gqk, blk, 0, stream>>>(
      xb, wkT, bk, nullptr, Kb, DINNER, DMODEL, DINNER, 0, 0, 0, 0);
  const dim3 gv(DINNER / BN, SEQLEN / BM, NBATCH);
  wmma_gemm<0, false, false, false, false, true><<<gv, blk, 0, stream>>>(
      xb, wvT, bv, nullptr, Vt, DINNER, DMODEL, /*ldo=*/SEQLEN,
      (long long)SEQLEN * DMODEL, 0, (long long)DINNER * SEQLEN, 0);

  // 3) masked scores S = tril(Q K^T): BT is simply K (row-major [m][e])
  const dim3 gsc(SEQLEN / BN, SEQLEN / BM, NBATCH);
  wmma_gemm<0, false, false, false, true, false><<<gsc, blk, 0, stream>>>(
      Qb, Kb, nullptr, nullptr, Sb, SEQLEN, DINNER, SEQLEN,
      (long long)SEQLEN * DINNER, (long long)SEQLEN * DINNER,
      (long long)SEQLEN * SEQLEN, 0);

  // 4) denominators (deterministic)
  const dim3 grs(SEQLEN, NBATCH);
  rowsum_kernel<<<grs, blk, 0, stream>>>(Sb, den);

  // 5) numerator with causal k-limit + row scaling: A = (S V) / den
  const dim3 gnum(DINNER / BN, SEQLEN / BM, NBATCH);
  wmma_gemm<0, false, true, true, false, false><<<gnum, blk, 0, stream>>>(
      Sb, Vt, nullptr, den, Ab, DINNER, SEQLEN, DINNER,
      (long long)SEQLEN * SEQLEN, (long long)DINNER * SEQLEN,
      (long long)SEQLEN * DINNER, (long long)SEQLEN);

  // 6) output projection: out = A Wo + bo (f32)
  const dim3 gout(DMODEL / BN, (unsigned)(MTOK / BM), 1);
  wmma_gemm<0, true, false, false, false, false><<<gout, blk, 0, stream>>>(
      Ab, woT, bo, nullptr, d_out, DMODEL, DINNER, DMODEL, 0, 0, 0, 0);
}
